// LongPrimitiveTransformer_31688268710632
// MI455X (gfx1250) — compile-verified, matrix operand order checked by reading, not run on hardware
//
#include <hip/hip_runtime.h>
#include <cmath>

// ---------------------------------------------------------------------------
// CDNA5 (gfx1250) implementation of the P4D long-primitive transformer.
// All dense matmuls go through v_wmma_f32_16x16x32_f16 (f32 HBM, f16 LDS tiles,
// f32 accumulate) with fused, compile-time-selected epilogues. Attention is
// flash-style (two passes, LDS bounce for the P-tile re-layout).
// ---------------------------------------------------------------------------

typedef __attribute__((ext_vector_type(16))) _Float16 v16h;
typedef __attribute__((ext_vector_type(8)))  _Float16 v8h;
typedef __attribute__((ext_vector_type(8)))  float    v8f;

#define DEV_INLINE __device__ __forceinline__

static constexpr int Bb = 2, Ll = 4, Nn = 6400, Fr = Bb * Ll;   // frames = B*L
static constexpr int HEADS = 4, DH = 256;                        // attention
static constexpr int QKVW = 3 * HEADS * DH;                      // 3072
static constexpr int INNER = HEADS * DH;                         // 1024

enum { EPI_NONE = 0, EPI_BIAS = 1, EPI_CBR = 2, EPI_BIAS_GELU = 3 };

DEV_INLINE v8f zero8() { v8f z = {0.f,0.f,0.f,0.f,0.f,0.f,0.f,0.f}; return z; }

DEV_INLINE v8f wmma_f16(v16h a, v16h b, v8f c) {
    return __builtin_amdgcn_wmma_f32_16x16x32_f16(false, a, false, b, (short)0, c, false, false);
}

// Fragment from an LDS tile row stored as f16 (row stride = LDSW halves).
// Per-lane pattern: halves [0..7] = K = kb..kb+7, [8..15] = kb+16..kb+23,
// kb = hi*8 (hi = lane>=16).  All accesses 16B aligned (LDSW=40 -> 80B rows).
static constexpr int LDSW = 40;
DEV_INLINE v16h frag_lds(const _Float16* __restrict__ p, int hi) {
    int kb = hi * 8;
    v8h lo = *(const v8h*)(p + kb);
    v8h up = *(const v8h*)(p + kb + 16);
    return __builtin_shufflevector(lo, up, 0,1,2,3,4,5,6,7,8,9,10,11,12,13,14,15);
}

// f32 fragment: 8 floats at p[0..7] and 8 at p[16..23], all float4-aligned.
// Works for global (global_load_b128) and LDS (ds_load_b128) pointers.
DEV_INLINE v16h frag_f32x4(const float* __restrict__ p) {
    float4 a0 = *(const float4*)(p);
    float4 a1 = *(const float4*)(p + 4);
    float4 b0 = *(const float4*)(p + 16);
    float4 b1 = *(const float4*)(p + 20);
    v16h v;
    v[0]  = (_Float16)a0.x; v[1]  = (_Float16)a0.y; v[2]  = (_Float16)a0.z; v[3]  = (_Float16)a0.w;
    v[4]  = (_Float16)a1.x; v[5]  = (_Float16)a1.y; v[6]  = (_Float16)a1.z; v[7]  = (_Float16)a1.w;
    v[8]  = (_Float16)b0.x; v[9]  = (_Float16)b0.y; v[10] = (_Float16)b0.z; v[11] = (_Float16)b0.w;
    v[12] = (_Float16)b1.x; v[13] = (_Float16)b1.y; v[14] = (_Float16)b1.z; v[15] = (_Float16)b1.w;
    return v;
}

// Strided fragment (V matrix): batch loads first so they pipeline, then convert.
DEV_INLINE v16h frag_strided(const float* __restrict__ p, size_t ld, int hi) {
    int kb = hi * 8;
    float t[16];
#pragma unroll
    for (int i = 0; i < 8; ++i) {
        t[i]     = p[(size_t)(kb + i) * ld];
        t[i + 8] = p[(size_t)(kb + 16 + i) * ld];
    }
    v16h v;
#pragma unroll
    for (int i = 0; i < 16; ++i) v[i] = (_Float16)t[i];
    return v;
}

// ---------------------------------------------------------------------------
// Fused WMMA GEMM:  Y = epi( X(R,K) @ W(N,K)^T ) [+ addend]
// EPI/HASADD are template params -> branch-free epilogue.
// block = 128 threads (4 waves); block tile 64x64; per-wave 16x64.
// A/B tiles staged in LDS as f16; fill is coalesced and branch-free
// (clamped indices + value select), with load/store phases split so all
// 32 global loads issue before the first wait.
// ---------------------------------------------------------------------------
template<int EPI, bool HASADD>
__global__ void __launch_bounds__(128)
k_gemm_wmma(const float* __restrict__ X, const float* __restrict__ W,
            const float* __restrict__ bias, const float* __restrict__ gain,
            const float* __restrict__ addend, float* __restrict__ Y,
            int Rows, int K, int N)
{
    __shared__ _Float16 sA[64 * LDSW];
    __shared__ _Float16 sB[64 * LDSW];

    int tid  = threadIdx.x;
    int wave = tid >> 5, lane = tid & 31;
    int lm = lane & 15, hi = lane >> 4;
    int m0blk = blockIdx.x * 64;
    int n0    = blockIdx.y * 64;
    int c  = tid & 31;          // column this thread fills (invariant)
    int r0 = tid >> 5;          // rows r0, r0+4, ..., r0+60

    v8f acc0 = zero8(), acc1 = zero8(), acc2 = zero8(), acc3 = zero8();

    for (int kk = 0; kk < K; kk += 32) {
        int ka = kk + c;
        int kc = (ka < K) ? ka : (K - 1);
        bool kv = ka < K;
        float va[16], vb[16];
        __syncthreads();
#pragma unroll
        for (int i = 0; i < 16; ++i) {          // load phase: 32 loads in flight
            int r = r0 + i * 4;
            int rc = m0blk + r; rc = (rc < Rows) ? rc : (Rows - 1);
            int bc = n0 + r;    bc = (bc < N)    ? bc : (N - 1);
            va[i] = X[(size_t)rc * K + kc];
            vb[i] = W[(size_t)bc * K + kc];
        }
        {   // speculative prefetch of next K-step (safe: dropped on fault)
            int rc = m0blk + r0; rc = (rc < Rows) ? rc : (Rows - 1);
            int bc = n0 + r0;    bc = (bc < N)    ? bc : (N - 1);
            __builtin_prefetch(X + (size_t)rc * K + kc + 32, 0, 1);
            __builtin_prefetch(W + (size_t)bc * K + kc + 32, 0, 1);
        }
#pragma unroll
        for (int i = 0; i < 16; ++i) {          // convert + store phase
            int r = r0 + i * 4;
            sA[r * LDSW + c] = (_Float16)(kv ? va[i] : 0.f);
            sB[r * LDSW + c] = (_Float16)(kv ? vb[i] : 0.f);
        }
        __syncthreads();

        v16h a = frag_lds(&sA[(wave * 16 + lm) * LDSW], hi);
        acc0 = wmma_f16(a, frag_lds(&sB[(lm)      * LDSW], hi), acc0);
        acc1 = wmma_f16(a, frag_lds(&sB[(16 + lm) * LDSW], hi), acc1);
        acc2 = wmma_f16(a, frag_lds(&sB[(32 + lm) * LDSW], hi), acc2);
        acc3 = wmma_f16(a, frag_lds(&sB[(48 + lm) * LDSW], hi), acc3);
    }

    int m0 = m0blk + wave * 16;
    v8f accs[4] = { acc0, acc1, acc2, acc3 };
#pragma unroll
    for (int t = 0; t < 4; ++t) {
#pragma unroll
        for (int j = 0; j < 8; ++j) {
            int m = m0 + j + hi * 8;
            int nn = n0 + t * 16 + lm;
            if (m < Rows && nn < N) {
                float s = accs[t][j];
                if constexpr (EPI == EPI_BIAS)      s += bias[nn];
                else if constexpr (EPI == EPI_CBR)  s = fmaxf(s * gain[nn] + bias[nn], 0.f);
                else if constexpr (EPI == EPI_BIAS_GELU) {
                    s += bias[nn];
                    s = 0.5f * s * (1.f + erff(s * 0.70710678118654752f));
                }
                if constexpr (HASADD) s += addend[(size_t)m * N + nn];
                Y[(size_t)m * N + nn] = s;
            }
        }
    }
}

// ---------------------------------------------------------------------------
// Flash attention (two pass).  QKV layout: token-major rows of 3072 =
// [q(1024)|k(1024)|v(1024)], head h at offset h*256.  n % 32 == 0 always.
// ---------------------------------------------------------------------------
DEV_INLINE void attn_s(const float* __restrict__ qkv, int s, int n, int h, int qt, int kb,
                       int lm, int hi, v8f& a0, v8f& a1)
{
    a0 = zero8(); a1 = zero8();
    const float* qp  = qkv + ((size_t)(s * n + qt * 16 + lm)) * QKVW + h * DH + hi * 8;
    const float* kp0 = qkv + ((size_t)(s * n + kb + lm)) * QKVW + INNER + h * DH + hi * 8;
    const float* kp1 = qkv + ((size_t)(s * n + kb + 16 + lm)) * QKVW + INNER + h * DH + hi * 8;
#pragma unroll
    for (int kk = 0; kk < DH; kk += 32) {
        v16h a = frag_f32x4(qp + kk);
        a0 = wmma_f16(a, frag_f32x4(kp0 + kk), a0);
        a1 = wmma_f16(a, frag_f32x4(kp1 + kk), a1);
    }
}

__global__ void k_attn_stats(const float* __restrict__ qkv, float* __restrict__ rmaxA,
                             float* __restrict__ rsumA, int n)
{
    int lane = threadIdx.x & 31, lm = lane & 15, hi = lane >> 4;
    int qt = blockIdx.x;
    int s = blockIdx.y / HEADS, h = blockIdx.y % HEADS;
    const float scale = 0.0625f;  // DIM_HEAD^-0.5
    float rmax[8], rsum[8];
#pragma unroll
    for (int j = 0; j < 8; ++j) { rmax[j] = -3.0e38f; rsum[j] = 0.f; }
    for (int kb = 0; kb < n; kb += 32) {
        v8f a0, a1; attn_s(qkv, s, n, h, qt, kb, lm, hi, a0, a1);
#pragma unroll
        for (int j = 0; j < 8; ++j) {
            float v0 = a0[j] * scale, v1 = a1[j] * scale;
            float mt = fmaxf(v0, v1);
            mt = fmaxf(mt, __shfl_xor(mt, 1)); mt = fmaxf(mt, __shfl_xor(mt, 2));
            mt = fmaxf(mt, __shfl_xor(mt, 4)); mt = fmaxf(mt, __shfl_xor(mt, 8));
            float nm = fmaxf(rmax[j], mt);
            float e = __expf(v0 - nm) + __expf(v1 - nm);
            e += __shfl_xor(e, 1); e += __shfl_xor(e, 2);
            e += __shfl_xor(e, 4); e += __shfl_xor(e, 8);
            rsum[j] = rsum[j] * __expf(rmax[j] - nm) + e;
            rmax[j] = nm;
        }
    }
    if (lm == 0) {
#pragma unroll
        for (int j = 0; j < 8; ++j) {
            int m = qt * 16 + j + hi * 8;
            rmaxA[(size_t)blockIdx.y * n + m] = rmax[j];
            rsumA[(size_t)blockIdx.y * n + m] = rsum[j];
        }
    }
}

__global__ void k_attn_out(const float* __restrict__ qkv, const float* __restrict__ rmaxA,
                           const float* __restrict__ rsumA, float* __restrict__ O, int n)
{
    __shared__ float sP[16 * 32];
    int lane = threadIdx.x & 31, lm = lane & 15, hi = lane >> 4;
    int qt = blockIdx.x;
    int s = blockIdx.y / HEADS, h = blockIdx.y % HEADS;
    int z = blockIdx.z;                       // 64-wide dim_head slice
    const float scale = 0.0625f;
    float rmax[8], rsum[8];
#pragma unroll
    for (int j = 0; j < 8; ++j) {
        int m = qt * 16 + j + hi * 8;
        rmax[j] = rmaxA[(size_t)blockIdx.y * n + m];
        rsum[j] = rsumA[(size_t)blockIdx.y * n + m];
    }
    v8f acc[4];
#pragma unroll
    for (int t = 0; t < 4; ++t) acc[t] = zero8();

    for (int kb = 0; kb < n; kb += 32) {
        v8f a0, a1; attn_s(qkv, s, n, h, qt, kb, lm, hi, a0, a1);
#pragma unroll
        for (int j = 0; j < 8; ++j) {          // D-layout -> LDS (m, kcol)
            int m = j + hi * 8;
            sP[m * 32 + lm]      = __expf(a0[j] * scale - rmax[j]);
            sP[m * 32 + 16 + lm] = __expf(a1[j] * scale - rmax[j]);
        }
        __syncthreads();
        // LDS -> A-layout fragment via aligned ds_load_b128s
        v16h aP = frag_f32x4(&sP[lm * 32 + hi * 8]);
#pragma unroll
        for (int t = 0; t < 4; ++t) {
            const float* vp = qkv + ((size_t)(s * n + kb)) * QKVW + 2 * INNER + h * DH
                              + z * 64 + t * 16 + lm;
            acc[t] = wmma_f16(aP, frag_strided(vp, QKVW, hi), acc[t]);
        }
        __syncthreads();
    }
#pragma unroll
    for (int t = 0; t < 4; ++t)
#pragma unroll
        for (int j = 0; j < 8; ++j) {
            int m = j + hi * 8;
            size_t row = (size_t)s * n + qt * 16 + m;
            O[row * INNER + h * DH + z * 64 + t * 16 + lm] = acc[t][j] / rsum[j];
        }
}

// ---------------------------------------------------------------------------
// Irregular / pointwise kernels
// ---------------------------------------------------------------------------
__global__ void k_fps(const float* __restrict__ xyz, int Np, int M, int* __restrict__ aidx)
{
    int f = blockIdx.x;
    const float* p = xyz + (size_t)f * Np * 3;
    __shared__ float dist[6400];
    __shared__ float sv[256];
    __shared__ int   si[256];
    __shared__ int   s_last;
    int t = threadIdx.x;
    for (int j = t; j < Np; j += 256) dist[j] = 1e10f;
    if (t == 0) { aidx[(size_t)f * M] = 0; s_last = 0; }
    __syncthreads();
    for (int i = 1; i < M; ++i) {
        int last = s_last;
        float lx = p[last * 3], ly = p[last * 3 + 1], lz = p[last * 3 + 2];
        float bv = -1.f; int bi = 0x7fffffff;
        for (int j = t; j < Np; j += 256) {
            float dx = p[j * 3] - lx, dy = p[j * 3 + 1] - ly, dz = p[j * 3 + 2] - lz;
            float nd = fminf(dist[j], dx * dx + dy * dy + dz * dz);
            dist[j] = nd;
            if (nd > bv || (nd == bv && j < bi)) { bv = nd; bi = j; }
        }
        sv[t] = bv; si[t] = bi;
        __syncthreads();
        for (int off = 128; off > 0; off >>= 1) {
            if (t < off) {
                float ov = sv[t + off]; int oi = si[t + off];
                if (ov > sv[t] || (ov == sv[t] && oi < si[t])) { sv[t] = ov; si[t] = oi; }
            }
            __syncthreads();
        }
        if (t == 0) { aidx[(size_t)f * M + i] = si[0]; s_last = si[0]; }
        __syncthreads();
    }
}

__global__ void k_gather_anchor(const float* __restrict__ xyz, const int* __restrict__ aidx,
                                float* __restrict__ anchors, int Np, int M)
{
    int g = blockIdx.x * blockDim.x + threadIdx.x;
    if (g >= Fr * M) return;
    int f = g / M;
    const float* s = xyz + ((size_t)f * Np + aidx[g]) * 3;
    anchors[(size_t)g * 3]     = s[0];
    anchors[(size_t)g * 3 + 1] = s[1];
    anchors[(size_t)g * 3 + 2] = s[2];
}

__global__ void k_ball_query(const float* __restrict__ anchors, const float* __restrict__ pts,
                             float r2, int Np, int M, int o, int* __restrict__ idx)
{
    int g = blockIdx.x * blockDim.x + threadIdx.x;
    if (g >= Fr * M) return;
    int f = g / M, b = f / Ll, l = f % Ll;
    int ls = min(max(l + o, 0), Ll - 1);
    const float* P = pts + (size_t)(b * Ll + ls) * Np * 3;
    float ax = anchors[(size_t)g * 3], ay = anchors[(size_t)g * 3 + 1], az = anchors[(size_t)g * 3 + 2];
    int* out = idx + (size_t)g * 9;
    int cnt = 0, first = 0;
    for (int j = 0; j < Np && cnt < 9; ++j) {
        float dx = P[j * 3] - ax, dy = P[j * 3 + 1] - ay, dz = P[j * 3 + 2] - az;
        if (dx * dx + dy * dy + dz * dz < r2) { if (!cnt) first = j; out[cnt++] = j; }
    }
    if (cnt == 0) { for (int kk = 0; kk < 9; ++kk) out[kk] = 0; }
    else          { for (int kk = cnt; kk < 9; ++kk) out[kk] = first; }
}

__global__ void k_build_xd(const float* __restrict__ anchors, const float* __restrict__ pts,
                           const int* __restrict__ idx, int Np, int M, int o,
                           float* __restrict__ Xd, int rows)
{
    int r = blockIdx.x * blockDim.x + threadIdx.x;
    if (r >= rows) return;
    int f = r / (M * 9), m = (r / 9) % M;
    int b = f / Ll, l = f % Ll;
    int ls = min(max(l + o, 0), Ll - 1);
    const float* P = pts + ((size_t)(b * Ll + ls) * Np + idx[r]) * 3;
    const float* A = anchors + ((size_t)f * M + m) * 3;
    Xd[(size_t)r * 4]     = P[0] - A[0];
    Xd[(size_t)r * 4 + 1] = P[1] - A[1];
    Xd[(size_t)r * 4 + 2] = P[2] - A[2];
    Xd[(size_t)r * 4 + 3] = (float)o;
}

__global__ void k_gather_feats(const float* __restrict__ feats, const int* __restrict__ idx,
                               int MK, int Np, int C, int o, float* __restrict__ Xf, int total)
{
    int g = blockIdx.x * blockDim.x + threadIdx.x;
    if (g >= total) return;
    int r = g / C, c = g % C;
    int f = r / MK, b = f / Ll, l = f % Ll;
    int ls = min(max(l + o, 0), Ll - 1);
    Xf[g] = feats[((size_t)(b * Ll + ls) * Np + idx[r]) * C + c];
}

__global__ void k_max_over_k(const float* __restrict__ H, float* __restrict__ acc,
                             int FM, int C, int accumulate)
{
    int g = blockIdx.x * blockDim.x + threadIdx.x;
    if (g >= FM * C) return;
    int fm = g / C, c = g % C;
    const float* p = H + (size_t)fm * 9 * C + c;
    float v = p[0];
#pragma unroll
    for (int kk = 1; kk < 9; ++kk) v = fmaxf(v, p[(size_t)kk * C]);
    if (accumulate) acc[g] += v; else acc[g] = v;
}

__global__ void k_nn3(const float* __restrict__ oxyz, const float* __restrict__ axyz,
                      int No, int Ma, int* __restrict__ widx, float* __restrict__ wgt)
{
    int g = blockIdx.x * blockDim.x + threadIdx.x;
    if (g >= Fr * No) return;
    int f = g / No;
    float qx = oxyz[(size_t)g * 3], qy = oxyz[(size_t)g * 3 + 1], qz = oxyz[(size_t)g * 3 + 2];
    const float* A = axyz + (size_t)f * Ma * 3;
    float d0 = 3e38f, d1 = 3e38f, d2 = 3e38f; int i0 = 0, i1 = 0, i2 = 0;
    for (int m = 0; m < Ma; ++m) {
        float dx = A[m * 3] - qx, dy = A[m * 3 + 1] - qy, dz = A[m * 3 + 2] - qz;
        float d = dx * dx + dy * dy + dz * dz;
        if (d < d0)      { d2 = d1; i2 = i1; d1 = d0; i1 = i0; d0 = d; i0 = m; }
        else if (d < d1) { d2 = d1; i2 = i1; d1 = d;  i1 = m; }
        else if (d < d2) { d2 = d;  i2 = m; }
    }
    float w0 = 1.f / (d0 + 1e-8f), w1 = 1.f / (d1 + 1e-8f), w2 = 1.f / (d2 + 1e-8f);
    float ws = w0 + w1 + w2;
    wgt[(size_t)g * 3] = w0 / ws; wgt[(size_t)g * 3 + 1] = w1 / ws; wgt[(size_t)g * 3 + 2] = w2 / ws;
    widx[(size_t)g * 3] = i0; widx[(size_t)g * 3 + 1] = i1; widx[(size_t)g * 3 + 2] = i2;
}

__global__ void k_interp_concat(const float* __restrict__ afeat, const float* __restrict__ ofeat,
                                const int* __restrict__ widx, const float* __restrict__ wgt,
                                int No, int Ma, int Ca, int Co, float* __restrict__ Xc, int total)
{
    int g = blockIdx.x * blockDim.x + threadIdx.x;
    if (g >= total) return;
    int Ct = Ca + Co;
    int r = g / Ct, c = g % Ct;
    if (c < Ca) {
        int f = r / No;
        const float* base = afeat + (size_t)f * Ma * Ca;
        Xc[g] = wgt[(size_t)r * 3]     * base[(size_t)widx[(size_t)r * 3]     * Ca + c]
              + wgt[(size_t)r * 3 + 1] * base[(size_t)widx[(size_t)r * 3 + 1] * Ca + c]
              + wgt[(size_t)r * 3 + 2] * base[(size_t)widx[(size_t)r * 3 + 2] * Ca + c];
    } else {
        Xc[g] = ofeat[(size_t)r * Co + (c - Ca)];
    }
}

__global__ void k_layernorm(const float* __restrict__ X, const float* __restrict__ g,
                            const float* __restrict__ b, float* __restrict__ Y, int rows)
{
    int wid = (blockIdx.x * blockDim.x + threadIdx.x) >> 5;
    int lane = threadIdx.x & 31;
    if (wid >= rows) return;
    float4 v = ((const float4*)(X + (size_t)wid * 128))[lane];
    float s = v.x + v.y + v.z + v.w;
    for (int m = 16; m >= 1; m >>= 1) s += __shfl_xor(s, m);
    float mu = s * (1.f / 128.f);
    float dx = v.x - mu, dy = v.y - mu, dz = v.z - mu, dw = v.w - mu;
    float q = dx * dx + dy * dy + dz * dz + dw * dw;
    for (int m = 16; m >= 1; m >>= 1) q += __shfl_xor(q, m);
    float rstd = rsqrtf(q * (1.f / 128.f) + 1e-5f);
    int c = lane * 4;
    float4 o;
    o.x = dx * rstd * g[c]     + b[c];
    o.y = dy * rstd * g[c + 1] + b[c + 1];
    o.z = dz * rstd * g[c + 2] + b[c + 2];
    o.w = dw * rstd * g[c + 3] + b[c + 3];
    ((float4*)(Y + (size_t)wid * 128))[lane] = o;
}

__global__ void k_transpose_rgb(const float* __restrict__ rgbs, float* __restrict__ feats0, int total)
{
    int g = blockIdx.x * blockDim.x + threadIdx.x;
    if (g >= total) return;
    int c = g % 3, rem = g / 3, npt = rem % Nn, f = rem / Nn;
    feats0[g] = rgbs[((size_t)f * 3 + c) * Nn + npt];
}

__global__ void k_relu(const float* __restrict__ X, float* __restrict__ Y, int total)
{
    int g = blockIdx.x * blockDim.x + threadIdx.x;
    if (g >= total) return;
    Y[g] = fmaxf(X[g], 0.f);
}

__global__ void k_prim_max(const float* __restrict__ pfX, float* __restrict__ prim, int total)
{
    int g = blockIdx.x * blockDim.x + threadIdx.x;     // Bb*800*128
    if (g >= total) return;
    int c = g % 128, srow = g / 128;                   // global seq 0..1599
    int b = srow / 800, q = srow % 800;
    float v = -3e38f;
#pragma unroll
    for (int t = 0; t < 32; ++t) v = fmaxf(v, pfX[((size_t)srow * 32 + t) * 128 + c]);
    prim[((size_t)b * 1600 + q) * 128 + c] = fmaxf(v, 0.f);   // relu(prim)
}

__global__ void k_copy_pe(const float* __restrict__ pe, float* __restrict__ prim, int total)
{
    int g = blockIdx.x * blockDim.x + threadIdx.x;     // Bb*800*128
    if (g >= total) return;
    int b = g / (800 * 128), rem = g % (800 * 128), q = rem / 128, c = rem % 128;
    prim[((size_t)b * 1600 + 800 + q) * 128 + c] = fmaxf(pe[g], 0.f);
}

__global__ void k_build_fused(const float* __restrict__ raw, const float* __restrict__ pfX,
                              const float* __restrict__ prim, float* __restrict__ fused, int total)
{
    int g = blockIdx.x * blockDim.x + threadIdx.x;     // Fr*Nn*384
    if (g >= total) return;
    int c = g % 384, row = g / 384;
    int f = row / Nn, npt = row % Nn;
    float v;
    if (c < 128)       v = raw[(size_t)row * 128 + c];
    else if (c < 256)  v = pfX[(size_t)row * 128 + (c - 128)];
    else {
        int b = f / Ll, l = f % Ll, p = npt / 32;
        v = prim[((size_t)b * 1600 + l * 200 + p) * 128 + (c - 256)];
    }
    fused[g] = v;
}

__global__ void k_out_transpose(const float* __restrict__ tmp, float* __restrict__ out, int total)
{
    int g = blockIdx.x * blockDim.x + threadIdx.x;     // Fr*Nn*12
    if (g >= total) return;
    int c = g % 12, row = g / 12, f = row / Nn, npt = row % Nn;
    out[((size_t)f * 12 + c) * Nn + npt] = tmp[g];
}

// ---------------------------------------------------------------------------
// Host orchestration
// ---------------------------------------------------------------------------
static inline int cdiv(int a, int b) { return (a + b - 1) / b; }

extern "C" void kernel_launch(void* const* d_in, const int* in_sizes, int n_in,
                              void* d_out, int out_size, void* d_ws, size_t ws_size,
                              hipStream_t stream)
{
    (void)in_sizes; (void)out_size; (void)ws_size;
    const float* xyzs = (const float*)d_in[0];   // (F, N, 3)
    const float* rgbs = (const float*)d_in[1];   // (F, 3, N)
    const float* pe   = (const float*)d_in[2];   // (B, 800, 128)

    // --- params flattened in jax tree order (dict keys sorted; 'W'<'b'<'g') ---
    int pi = 3;
    auto P = [&]() -> const float* {
        int i = (pi < n_in) ? pi : (n_in - 1); ++pi; return (const float*)d_in[i];
    };
    struct CBN { const float *W, *b, *g; int ci, co; };
    auto getcbn = [&](int ci, int co) { CBN c; c.W = P(); c.b = P(); c.g = P(); c.ci = ci; c.co = co; return c; };
    struct LIN { const float *W, *b; };
    auto getlin = [&](bool bias) { LIN l; l.W = P(); l.b = bias ? P() : nullptr; return l; };

    CBN c1d = getcbn(4, 32),   c1f = getcbn(3, 32),    c1m0 = getcbn(32, 64),    c1m1 = getcbn(64, 128);
    CBN c2d = getcbn(4, 128),  c2f = getcbn(128, 128), c2m0 = getcbn(128, 128),  c2m1 = getcbn(128, 256);
    CBN c3d = getcbn(4, 256),  c3f = getcbn(256, 256), c3m0 = getcbn(256, 256),  c3m1 = getcbn(256, 512);
    CBN c4d = getcbn(4, 512),  c4f = getcbn(512, 512), c4m0 = getcbn(512, 512),  c4m1 = getcbn(512, 1024);
    CBN d1m0 = getcbn(131, 128),  d1m1 = getcbn(128, 128);
    CBN d2m0 = getcbn(384, 128),  d2m1 = getcbn(128, 128);
    CBN d3m0 = getcbn(512, 256),  d3m1 = getcbn(256, 256);
    CBN d4m0 = getcbn(1536, 256), d4m1 = getcbn(256, 256);
    LIN out1 = getlin(true), out2 = getlin(true);
    struct TFL { LIN ff1, ff2; const float *ln1b, *ln1g, *ln2b, *ln2g; LIN out; const float* qkvW; };
    auto gettf = [&]() { TFL t; t.ff1 = getlin(true); t.ff2 = getlin(true);
                         t.ln1b = P(); t.ln1g = P(); t.ln2b = P(); t.ln2g = P();
                         t.out = getlin(true); t.qkvW = P(); return t; };
    TFL tfa[2] = { gettf(), gettf() };
    TFL tfb[2] = { gettf(), gettf() };

    // --- workspace bump allocator ---
    char* wsb = (char*)d_ws;
    size_t cur = 0;
    auto alloc = [&](size_t bytes) -> char* { char* p = wsb + cur; cur += (bytes + 255) & ~(size_t)255; return p; };
    auto allocf = [&](size_t elems) -> float* { return (float*)alloc(elems * 4); };
    auto alloci = [&](size_t elems) -> int*   { return (int*)alloc(elems * 4); };

    // persistent buffers
    float* feats0 = allocf((size_t)Fr * Nn * 3);
    float* x1 = allocf((size_t)Fr * 1600 * 3);
    float* x2 = allocf((size_t)Fr * 400 * 3);
    float* x3 = allocf((size_t)Fr * 100 * 3);
    float* x4 = allocf((size_t)Fr * 50 * 3);
    float* f1 = allocf((size_t)Fr * 1600 * 128);
    float* f2 = allocf((size_t)Fr * 400 * 256);
    float* f3 = allocf((size_t)Fr * 100 * 512);
    float* f4 = allocf((size_t)Fr * 50 * 1024);
    float* fd4 = allocf((size_t)Fr * 100 * 256);
    float* fd3 = allocf((size_t)Fr * 400 * 256);
    float* fd2 = allocf((size_t)Fr * 1600 * 128);
    float* raw = allocf((size_t)Fr * Nn * 128);
    float* pfX = allocf((size_t)Fr * Nn * 128);
    float* prim = allocf((size_t)Bb * 1600 * 128);

    auto gemm = [&](const float* X, const float* W, const float* bias, const float* gain,
                    const float* add, float* Y, int R, int K, int N, int epi) {
        dim3 g(cdiv(R, 64), cdiv(N, 64)), b(128);
        if (add) {
            if (epi == EPI_BIAS)
                k_gemm_wmma<EPI_BIAS, true><<<g, b, 0, stream>>>(X, W, bias, gain, add, Y, R, K, N);
            else
                k_gemm_wmma<EPI_CBR, true><<<g, b, 0, stream>>>(X, W, bias, gain, add, Y, R, K, N);
        } else {
            switch (epi) {
            case EPI_NONE:      k_gemm_wmma<EPI_NONE,      false><<<g, b, 0, stream>>>(X, W, bias, gain, nullptr, Y, R, K, N); break;
            case EPI_BIAS:      k_gemm_wmma<EPI_BIAS,      false><<<g, b, 0, stream>>>(X, W, bias, gain, nullptr, Y, R, K, N); break;
            case EPI_CBR:       k_gemm_wmma<EPI_CBR,       false><<<g, b, 0, stream>>>(X, W, bias, gain, nullptr, Y, R, K, N); break;
            default:            k_gemm_wmma<EPI_BIAS_GELU, false><<<g, b, 0, stream>>>(X, W, bias, gain, nullptr, Y, R, K, N); break;
            }
        }
    };
    auto launch1 = [&](int total) { return dim3(cdiv(total, 256)); };

    // --- feats0 = rgbs transposed to point-major ---
    { int tot = Fr * Nn * 3; k_transpose_rgb<<<launch1(tot), 256, 0, stream>>>(rgbs, feats0, tot); }

    // --- P4D conv level ---
    auto run_conv = [&](const float* xyzL, const float* featL, int Np, int Cin, int M,
                        const CBN& pd, const CBN& pf, const CBN& pm0, const CBN& pm1,
                        float r, int tr, float* outXyz, float* outFeat) {
        size_t save = cur;
        int rows = Fr * M * 9;
        int* aidx = alloci((size_t)Fr * M);
        int* idx  = alloci((size_t)rows);
        float* Xd = allocf((size_t)rows * 4);
        float* Xf = allocf((size_t)rows * Cin);
        float* Hd = allocf((size_t)rows * pd.co);
        int cmax = pf.co; if (pm0.co > cmax) cmax = pm0.co; if (pm1.co > cmax) cmax = pm1.co;
        float* Hp = allocf((size_t)rows * cmax);
        float* Hq = allocf((size_t)rows * cmax);

        k_fps<<<Fr, 256, 0, stream>>>(xyzL, Np, M, aidx);
        k_gather_anchor<<<launch1(Fr * M), 256, 0, stream>>>(xyzL, aidx, outXyz, Np, M);
        for (int o = -tr; o <= tr; ++o) {
            k_ball_query<<<launch1(Fr * M), 256, 0, stream>>>(outXyz, xyzL, r * r, Np, M, o, idx);
            k_build_xd<<<launch1(rows), 256, 0, stream>>>(outXyz, xyzL, idx, Np, M, o, Xd, rows);
            k_gather_feats<<<launch1(rows * Cin), 256, 0, stream>>>(featL, idx, M * 9, Np, Cin, o, Xf, rows * Cin);
            gemm(Xd, pd.W, pd.b, pd.g, nullptr, Hd, rows, 4,      pd.co,  EPI_CBR);
            gemm(Xf, pf.W, pf.b, pf.g, Hd,      Hp, rows, Cin,    pf.co,  EPI_CBR);
            gemm(Hp, pm0.W, pm0.b, pm0.g, nullptr, Hq, rows, pf.co,  pm0.co, EPI_CBR);
            gemm(Hq, pm1.W, pm1.b, pm1.g, nullptr, Hp, rows, pm0.co, pm1.co, EPI_CBR);
            k_max_over_k<<<launch1(Fr * M * pm1.co), 256, 0, stream>>>(Hp, outFeat, Fr * M, pm1.co, o != -tr);
        }
        cur = save;
    };

    run_conv(xyzs, feats0, Nn,   3,   1600, c1d, c1f, c1m0, c1m1, 0.9f, 0, x1, f1);
    run_conv(x1,   f1,     1600, 128, 400,  c2d, c2f, c2m0, c2m1, 1.8f, 0, x2, f2);
    run_conv(x2,   f2,     400,  256, 100,  c3d, c3f, c3m0, c3m1, 3.6f, 1, x3, f3);
    run_conv(x3,   f3,     100,  512, 50,   c4d, c4f, c4m0, c4m1, 7.2f, 0, x4, f4);

    // --- feature propagation (3-NN interpolation + MLP) ---
    auto run_deconv = [&](const float* axyz, int Ma, const float* oxyz, int No,
                          const float* afeat, int Ca, const float* ofeat, int Co,
                          const CBN& m0, const CBN& m1, float* outFeat) {
        size_t save = cur;
        int rows = Fr * No;
        int* widx = alloci((size_t)rows * 3);
        float* wgt = allocf((size_t)rows * 3);
        float* Xc = allocf((size_t)rows * (Ca + Co));
        float* Ht = allocf((size_t)rows * m0.co);
        k_nn3<<<launch1(rows), 256, 0, stream>>>(oxyz, axyz, No, Ma, widx, wgt);
        int tot = rows * (Ca + Co);
        k_interp_concat<<<launch1(tot), 256, 0, stream>>>(afeat, ofeat, widx, wgt, No, Ma, Ca, Co, Xc, tot);
        gemm(Xc, m0.W, m0.b, m0.g, nullptr, Ht,      rows, Ca + Co, m0.co, EPI_CBR);
        gemm(Ht, m1.W, m1.b, m1.g, nullptr, outFeat, rows, m0.co,   m1.co, EPI_CBR);
        cur = save;
    };

    run_deconv(x4, 50,   x3,   100,  f4,  1024, f3,     512, d4m0, d4m1, fd4);
    run_deconv(x3, 100,  x2,   400,  fd4, 256,  f2,     256, d3m0, d3m1, fd3);
    run_deconv(x2, 400,  x1,   1600, fd3, 256,  f1,     128, d2m0, d2m1, fd2);
    run_deconv(x1, 1600, xyzs, Nn,   fd2, 128,  feats0, 3,   d1m0, d1m1, raw);

    // --- transformer driver (chunked over sequences) ---
    auto run_tf = [&](const TFL* Ls, float* Xst, int nseq, int n, int chunk) {
        size_t save = cur;
        int tokens = nseq * n;
        int ctok = chunk * n;
        float* LN   = allocf((size_t)tokens * 128);
        float* FF   = allocf((size_t)tokens * 128);
        float* QKVb = allocf((size_t)ctok * QKVW);
        float* Ob   = allocf((size_t)ctok * INNER);
        float* rmaxA = allocf((size_t)chunk * HEADS * n);
        float* rsumA = allocf((size_t)chunk * HEADS * n);
        for (int li = 0; li < 2; ++li) {
            const TFL& t = Ls[li];
            k_layernorm<<<cdiv(tokens, 8), 256, 0, stream>>>(Xst, t.ln1g, t.ln1b, LN, tokens);
            for (int s0 = 0; s0 < nseq; s0 += chunk) {
                int cs = (nseq - s0 < chunk) ? (nseq - s0) : chunk;
                int ct = cs * n;
                size_t r0 = (size_t)s0 * n;
                gemm(LN + r0 * 128, t.qkvW, nullptr, nullptr, nullptr, QKVb, ct, 128, QKVW, EPI_NONE);
                k_attn_stats<<<dim3(n / 16, cs * HEADS), 32, 0, stream>>>(QKVb, rmaxA, rsumA, n);
                k_attn_out<<<dim3(n / 16, cs * HEADS, 4), 32, 0, stream>>>(QKVb, rmaxA, rsumA, Ob, n);
                gemm(Ob, t.out.W, t.out.b, nullptr, Xst + r0 * 128, Xst + r0 * 128, ct, INNER, 128, EPI_BIAS);
            }
            k_layernorm<<<cdiv(tokens, 8), 256, 0, stream>>>(Xst, t.ln2g, t.ln2b, LN, tokens);
            gemm(LN, t.ff1.W, t.ff1.b, nullptr, nullptr, FF, tokens, 128, 128, EPI_BIAS_GELU);
            gemm(FF, t.ff2.W, t.ff2.b, nullptr, Xst, Xst, tokens, 128, 128, EPI_BIAS);
        }
        cur = save;
    };

    // tf1: 1600 sequences of 32 tokens
    { int tot = Fr * Nn * 128; k_relu<<<launch1(tot), 256, 0, stream>>>(raw, pfX, tot); }
    run_tf(tfa, pfX, 1600, 32, 100);

    // primitive tokens: max over 32 points (+relu), then relu(pe), then tf2
    { int tot = Bb * 800 * 128; k_prim_max<<<launch1(tot), 256, 0, stream>>>(pfX, prim, tot); }
    { int tot = Bb * 800 * 128; k_copy_pe<<<launch1(tot), 256, 0, stream>>>(pe, prim, tot); }
    run_tf(tfb, prim, 2, 1600, 2);

    // --- fused head ---
    {
        size_t save = cur;
        float* fused = allocf((size_t)Fr * Nn * 384);
        float* h1    = allocf((size_t)Fr * Nn * 128);
        float* t12   = allocf((size_t)Fr * Nn * 12);
        int tot = Fr * Nn * 384;
        k_build_fused<<<launch1(tot), 256, 0, stream>>>(raw, pfX, prim, fused, tot);
        gemm(fused, out1.W, out1.b, nullptr, nullptr, h1,  Fr * Nn, 384, 128, EPI_BIAS);
        gemm(h1,    out2.W, out2.b, nullptr, nullptr, t12, Fr * Nn, 128, 12,  EPI_BIAS);
        int tot2 = Fr * Nn * 12;
        k_out_transpose<<<launch1(tot2), 256, 0, stream>>>(t12, (float*)d_out, tot2);
        cur = save;
    }
}
